// CausalFrameLinearAttention_86955907875477
// MI455X (gfx1250) — matrix operand order repaired and checked by reading, not compiled, not run on hardware
//
#include <hip/hip_runtime.h>

// ---------------- constants ----------------
constexpr int nB = 8, nCh = 48, nH = 4, nE = 12, nF = 65, nT = 1024;
constexpr int nChunk = 16, L = 64, D = 780, Dp = 800, BH = 32;
constexpr int Ksteps = Dp / 32;   // 25 k-steps of 32 for bf16 WMMA
constexpr int Ntiles = Dp / 16;   // 50 n-tiles of 16
constexpr float EPSv = 1e-5f;

typedef __bf16 bf16;
typedef __attribute__((ext_vector_type(16))) __bf16 v16bf;
typedef __attribute__((ext_vector_type(8)))  float  v8f;

// ---------------- bf16 helpers ----------------
static __device__ inline bf16 f2bf(float f) {
  unsigned u; __builtin_memcpy(&u, &f, 4);
  unsigned r = u + 0x7fffu + ((u >> 16) & 1u);
  unsigned short s = (unsigned short)(r >> 16);
  bf16 b; __builtin_memcpy(&b, &s, 2); return b;
}
static __device__ inline float bf2f(bf16 b) {
  unsigned short s; __builtin_memcpy(&s, &b, 2);
  unsigned u = ((unsigned)s) << 16;
  float f; __builtin_memcpy(&f, &u, 4); return f;
}

struct Frag16 { uint4 lo, hi; };
static __device__ inline v16bf asv16(Frag16 f) { v16bf v; __builtin_memcpy(&v, &f, 32); return v; }

// A fragment 16x32 (MxK), row-major source (K contiguous), base at (m0,k0).
// ISA layout: lanes 0-15 hold K=[0..8)+[16..24), lanes 16-31 hold K=[8..16)+[24..32).
static __device__ inline v16bf load_a_rm(const bf16* base, int ld, int lane) {
  int m = lane & 15, g = lane >> 4;
  const bf16* p = base + m * ld + 8 * g;
  Frag16 f;
  f.lo = *reinterpret_cast<const uint4*>(p);        // elements 0..7
  f.hi = *reinterpret_cast<const uint4*>(p + 16);   // elements 8..15
  return asv16(f);
}
// B fragment 32x16 (KxN), column-major source (K contiguous per column): src[n*ld + k].
// ISA layout: lanes 0-15 hold K=[0..8)+[16..24), lanes 16-31 hold K=[8..16)+[24..32), N = lane%16.
static __device__ inline v16bf load_b_cm(const bf16* base, int ld, int lane) {
  int n = lane & 15, g = lane >> 4;
  const bf16* p = base + n * ld + 8 * g;
  Frag16 f;
  f.lo = *reinterpret_cast<const uint4*>(p);
  f.hi = *reinterpret_cast<const uint4*>(p + 16);
  return asv16(f);
}

static __device__ inline v8f wmma_bf16(v16bf a, v16bf b, v8f c) {
  return __builtin_amdgcn_wmma_f32_16x16x32_bf16(false, a, false, b, (short)0, c, false, false);
}

// ---- CDNA5 async global->LDS copy (ASYNCcnt-tracked, ISA ch.10 / 15.18) ----
static __device__ inline void async_g2l_b128(unsigned lds_off, const bf16* gbase, unsigned byte_off) {
  asm volatile("global_load_async_to_lds_b128 %0, %1, %2"
               :: "v"(lds_off), "v"(byte_off), "s"(gbase)
               : "memory");
}
static __device__ inline void wait_async0() {
  asm volatile("s_wait_asynccnt 0x0" ::: "memory");
}

// ---------------- stage 1: QKV proj + PReLU + LN + feature map ----------------
// Writes: Qf,Kf row-major [bh][t][d] bf16; Ktg,Vtg chunk-transposed [bh][chunk][d][l] bf16.
__global__ __launch_bounds__(256) void qkv_kernel(
    const float* __restrict__ x,
    const float* __restrict__ Wq, const float* __restrict__ bq, const float* __restrict__ aq,
    const float* __restrict__ gq, const float* __restrict__ zq,
    const float* __restrict__ Wk, const float* __restrict__ bk, const float* __restrict__ ak,
    const float* __restrict__ gk, const float* __restrict__ zk,
    const float* __restrict__ Wv, const float* __restrict__ bv, const float* __restrict__ av,
    const float* __restrict__ gv, const float* __restrict__ zv,
    unsigned short* __restrict__ Qf, unsigned short* __restrict__ Kf,
    unsigned short* __restrict__ Ktg, unsigned short* __restrict__ Vtg)
{
  __shared__ float Wl[3][nCh * nCh];
  __shared__ float bl[3][nCh], gl[3][nCh], zl[3][nCh], al[3][nH];
  const float* Ws[3] = {Wq, Wk, Wv};
  const float* bs[3] = {bq, bk, bv};
  const float* gs[3] = {gq, gk, gv};
  const float* zs[3] = {zq, zk, zv};
  const float* as[3] = {aq, ak, av};
  int tid = threadIdx.x;
  for (int p = 0; p < 3; ++p) {
    for (int i = tid; i < nCh * nCh; i += 256) Wl[p][i] = Ws[p][i];
    if (tid < nCh) { bl[p][tid] = bs[p][tid]; gl[p][tid] = gs[p][tid]; zl[p][tid] = zs[p][tid]; }
    if (tid < nH) al[p][tid] = as[p][tid];
  }
  __syncthreads();

  int t = blockIdx.x * 256 + tid;
  int f = blockIdx.y;
  int b = blockIdx.z;

  float xin[nCh];
#pragma unroll
  for (int c = 0; c < nCh; ++c)
    xin[c] = x[(((size_t)b * nCh + c) * nF + f) * nT + t];

  for (int p = 0; p < 3; ++p) {
    for (int h = 0; h < nH; ++h) {
      float y[nE];
#pragma unroll
      for (int e = 0; e < nE; ++e) {
        int o = h * nE + e;
        float s = bl[p][o];
#pragma unroll
        for (int c = 0; c < nCh; ++c) s += Wl[p][o * nCh + c] * xin[c];
        y[e] = (s >= 0.f) ? s : al[p][h] * s;          // per-head PReLU
      }
      float mu = 0.f;
#pragma unroll
      for (int e = 0; e < nE; ++e) mu += y[e];
      mu *= (1.f / nE);
      float var = 0.f;
#pragma unroll
      for (int e = 0; e < nE; ++e) { float d = y[e] - mu; var += d * d; }
      var *= (1.f / nE);
      float inv = rsqrtf(var + EPSv);
      size_t baseRM = ((size_t)(b * nH + h) * nT + t) * Dp;
      size_t baseTR = (((size_t)(b * nH + h) * nChunk + (t >> 6)) * Dp) * L + (t & 63);
#pragma unroll
      for (int e = 0; e < nE; ++e) {
        float v = (y[e] - mu) * inv * gl[p][h * nE + e] + zl[p][h * nE + e];
        if (p < 2) v = (v > 0.f) ? (v + 1.f) : __expf(v);  // elu(v)+1 feature map for Q,K
        bf16 bv16 = f2bf(v);
        int d = e * nF + f;
        if (p == 0) ((bf16*)Qf)[baseRM + d] = bv16;
        if (p == 1) { ((bf16*)Kf)[baseRM + d] = bv16; ((bf16*)Ktg)[baseTR + (size_t)d * L] = bv16; }
        if (p == 2) ((bf16*)Vtg)[baseTR + (size_t)d * L] = bv16;
      }
    }
  }
  // zero the padding rows/cols D..Dp
  if (f == 0) {
    bf16 z0 = f2bf(0.f);
    for (int h = 0; h < nH; ++h) {
      size_t baseRM = ((size_t)(b * nH + h) * nT + t) * Dp;
      size_t baseTR = (((size_t)(b * nH + h) * nChunk + (t >> 6)) * Dp) * L + (t & 63);
      for (int d = D; d < Dp; ++d) {
        ((bf16*)Qf)[baseRM + d] = z0;
        ((bf16*)Kf)[baseRM + d] = z0;
        ((bf16*)Ktg)[baseTR + (size_t)d * L] = z0;
        ((bf16*)Vtg)[baseTR + (size_t)d * L] = z0;
      }
    }
  }
}

// ---------------- stage 2: chunked causal linear attention ----------------
constexpr int LDS_KT = 0;                      // Kt[Dp][L]  bf16  (102400 B)
constexpr int LDS_VT = Dp * L * 2;             // Vt[Dp][L]  bf16  (102400 B)
constexpr int LDS_AB = 2 * Dp * L * 2;         // Abf[L][L]  bf16  (8192 B)
constexpr int LDS_RS = LDS_AB + L * L * 2;     // rsum[L]    f32
constexpr int LDS_RD = LDS_RS + L * 4;         // rden[L]    f32
constexpr int LDS_BYTES = LDS_RD + L * 4;      // 213504 B of the 320 KB WGP LDS

__global__ __launch_bounds__(256) void attn_kernel(
    const unsigned short* __restrict__ Qfu, const unsigned short* __restrict__ Kfu,
    const unsigned short* __restrict__ Ktgu, const unsigned short* __restrict__ Vtgu,
    float* __restrict__ S, unsigned short* __restrict__ Sbfu,
    float* __restrict__ Z, float* __restrict__ O)
{
  extern __shared__ __align__(16) char lds[];
  bf16*  Kt   = (bf16*)(lds + LDS_KT);
  bf16*  Vt   = (bf16*)(lds + LDS_VT);
  bf16*  Abf  = (bf16*)(lds + LDS_AB);
  float* rsum = (float*)(lds + LDS_RS);
  float* rden = (float*)(lds + LDS_RD);

  int bh = blockIdx.x, tid = threadIdx.x;
  int lane = tid & 31, wv = tid >> 5;

  const bf16* Qb  = (const bf16*)Qfu  + (size_t)bh * nT * Dp;
  const bf16* Kb  = (const bf16*)Kfu  + (size_t)bh * nT * Dp;
  const bf16* Ktg = (const bf16*)Ktgu + (size_t)bh * nChunk * Dp * L;
  const bf16* Vtg = (const bf16*)Vtgu + (size_t)bh * nChunk * Dp * L;
  float* Sb   = S + (size_t)bh * Dp * Dp;          // fp32 master, transposed: S_T[v][d]
  bf16*  Sbfb = (bf16*)Sbfu + (size_t)bh * Dp * Dp; // bf16 shadow, same layout
  float* zb = Z + (size_t)bh * Dp;
  float* Ob = O + (size_t)bh * nT * Dp;

  // zero running state (ws is poisoned; must not rely on leftover state)
  for (int i = tid; i < Dp * Dp / 4; i += 256)
    reinterpret_cast<float4*>(Sb)[i] = make_float4(0.f, 0.f, 0.f, 0.f);
  for (int i = tid; i < Dp * Dp / 8; i += 256)
    reinterpret_cast<uint4*>(Sbfb)[i] = make_uint4(0u, 0u, 0u, 0u);
  for (int d = tid; d < Dp; d += 256) zb[d] = 0.f;
  __threadfence();
  __syncthreads();

  unsigned ktBase = (unsigned)(uintptr_t)(void*)Kt;   // LDS byte offsets for async DMA
  unsigned vtBase = (unsigned)(uintptr_t)(void*)Vt;

  for (int nc = 0; nc < nChunk; ++nc) {
    const bf16* Qc = Qb + (size_t)nc * L * Dp;
    const bf16* Kc = Kb + (size_t)nc * L * Dp;
    const bf16* Ks = Ktg + (size_t)nc * Dp * L;   // chunk-transposed, contiguous 100 KB
    const bf16* Vs = Vtg + (size_t)nc * Dp * L;

    // async-stage K^T,V^T chunk tiles into LDS (ASYNCcnt path), 16 B per lane per issue
    for (int i = tid; i < (Dp * L * 2) / 16; i += 256) {
      async_g2l_b128(ktBase + i * 16, Ks, i * 16);
      async_g2l_b128(vtBase + i * 16, Vs, i * 16);
    }
    // prefetch next chunk's streaming operands into GL2 while DMA runs
    if (nc + 1 < nChunk) {
      const char* pq = (const char*)(Qc + (size_t)L * Dp);
      const char* pk = (const char*)(Kc + (size_t)L * Dp);
      for (int i = tid; i < (L * Dp * 2) / 256; i += 256) {
        __builtin_prefetch(pq + i * 256, 0, 1);
        __builtin_prefetch(pk + i * 256, 0, 1);
      }
    }
    if (tid < L) rsum[tid] = 0.f;
    wait_async0();
    __syncthreads();

    // A = tril(Qc * Kc^T)   (16 tiles, 25 k-steps each)
    for (int tt = wv; tt < 16; tt += 8) {
      int mt = tt >> 2, nt = tt & 3;
      v8f acc;
#pragma unroll
      for (int i = 0; i < 8; ++i) acc[i] = 0.f;
      for (int kk = 0; kk < Ksteps; ++kk) {
        v16bf a = load_a_rm(Qc + mt * 16 * Dp + kk * 32, Dp, lane);
        v16bf b = load_b_cm(Kc + nt * 16 * Dp + kk * 32, Dp, lane);  // row-major K == col-major (d,l)
        acc = wmma_bf16(a, b, acc);
      }
      int g = lane >> 4, nn = lane & 15;
#pragma unroll
      for (int r = 0; r < 8; ++r) {
        int l = mt * 16 + r + 8 * g, l2 = nt * 16 + nn;
        float val = (l2 <= l) ? acc[r] : 0.f;                         // causal mask
        Abf[l * L + l2] = f2bf(val);
        atomicAdd(&rsum[l], val);
      }
    }
    __syncthreads();

    // den = Q.z + rowsum(A) + eps ; keep reciprocal
    if (tid < L) {
      const bf16* qr = Qc + (size_t)tid * Dp;
      float qz = 0.f;
      for (int d = 0; d < Dp; ++d) qz += bf2f(qr[d]) * zb[d];
      rden[tid] = 1.f / (qz + rsum[tid] + EPSv);
    }
    __syncthreads();

    // o = (Q*S + A*V) / den   (4 x 50 tiles); S read from bf16 shadow (fast b128 path)
    for (int tt = wv; tt < 4 * Ntiles; tt += 8) {
      int mt = tt / Ntiles, vt = tt - mt * Ntiles;
      v8f acc;
#pragma unroll
      for (int i = 0; i < 8; ++i) acc[i] = 0.f;
      for (int kk = 0; kk < Ksteps; ++kk) {
        v16bf a = load_a_rm(Qc + mt * 16 * Dp + kk * 32, Dp, lane);
        v16bf b = load_b_cm(Sbfb + (size_t)vt * 16 * Dp + kk * 32, Dp, lane);
        acc = wmma_bf16(a, b, acc);
      }
#pragma unroll
      for (int k2 = 0; k2 < 2; ++k2) {
        v16bf a = load_a_rm(Abf + mt * 16 * L + k2 * 32, L, lane);
        v16bf b = load_b_cm(Vt + vt * 16 * L + k2 * 32, L, lane);
        acc = wmma_bf16(a, b, acc);
      }
      int g = lane >> 4, nn = lane & 15;
#pragma unroll
      for (int r = 0; r < 8; ++r) {
        int l = mt * 16 + r + 8 * g;
        Ob[(size_t)(nc * L + l) * Dp + vt * 16 + nn] = acc[r] * rden[l];
      }
    }
    __syncthreads();   // all reads of S must finish before S update

    // S_T[v][d] += sum_l K[l][d]*V[l][v]  (50x50 tiles): fp32 RMW + bf16 shadow store
    for (int tt = wv; tt < Ntiles * Ntiles; tt += 8) {
      int dt = tt / Ntiles, vt = tt - dt * Ntiles;
      int g = lane >> 4, nn = lane & 15;
      float* cp = Sb + (size_t)(vt * 16 + nn) * Dp + dt * 16 + 8 * g;
      float4 c0 = *reinterpret_cast<float4*>(cp);
      float4 c1 = *reinterpret_cast<float4*>(cp + 4);
      v8f acc;
      acc[0] = c0.x; acc[1] = c0.y; acc[2] = c0.z; acc[3] = c0.w;
      acc[4] = c1.x; acc[5] = c1.y; acc[6] = c1.z; acc[7] = c1.w;
#pragma unroll
      for (int k2 = 0; k2 < 2; ++k2) {
        v16bf a = load_a_rm(Kt + dt * 16 * L + k2 * 32, L, lane);   // K^T fragment from LDS
        v16bf b = load_b_cm(Vt + vt * 16 * L + k2 * 32, L, lane);
        acc = wmma_bf16(a, b, acc);
      }
      *reinterpret_cast<float4*>(cp)     = make_float4(acc[0], acc[1], acc[2], acc[3]);
      *reinterpret_cast<float4*>(cp + 4) = make_float4(acc[4], acc[5], acc[6], acc[7]);
      bf16 pk[8];
#pragma unroll
      for (int r = 0; r < 8; ++r) pk[r] = f2bf(acc[r]);
      bf16* sp = Sbfb + (size_t)(vt * 16 + nn) * Dp + dt * 16 + 8 * g;
      *reinterpret_cast<uint4*>(sp) = *reinterpret_cast<const uint4*>(pk);  // one 16 B store
    }
    for (int d = tid; d < Dp; d += 256) {
      float s = 0.f;
      for (int l = 0; l < L; ++l) s += bf2f(Kt[d * L + l]);
      zb[d] += s;
    }
    __threadfence();
    __syncthreads();
  }
}

// ---------------- stage 3: output proj + PReLU + channel LN + residual ----------------
__global__ __launch_bounds__(256) void proj_kernel(
    const float* __restrict__ O, const float* __restrict__ x,
    const float* __restrict__ Wp, const float* __restrict__ bp, const float* __restrict__ ap,
    const float* __restrict__ gp, const float* __restrict__ zp,
    float* __restrict__ out)
{
  __shared__ float Wl[nCh * nCh], bl[nCh], gl[nCh], zl[nCh];
  __shared__ float al;
  int tid = threadIdx.x;
  for (int i = tid; i < nCh * nCh; i += 256) Wl[i] = Wp[i];
  if (tid < nCh) { bl[tid] = bp[tid]; gl[tid] = gp[tid]; zl[tid] = zp[tid]; }
  if (tid == 0) al = ap[0];
  __syncthreads();

  int t = blockIdx.x * 256 + tid, f = blockIdx.y, b = blockIdx.z;
  float att[nCh];
#pragma unroll
  for (int c = 0; c < nCh; ++c) {
    int h = c / nE, e = c - h * nE;
    att[c] = O[((size_t)(b * nH + h) * nT + t) * Dp + (size_t)e * nF + f];
  }
  float y[nCh];
  float mu = 0.f;
#pragma unroll
  for (int o = 0; o < nCh; ++o) {
    float s = bl[o];
#pragma unroll
    for (int c = 0; c < nCh; ++c) s += Wl[o * nCh + c] * att[c];
    s = (s >= 0.f) ? s : al * s;
    y[o] = s; mu += s;
  }
  mu *= (1.f / nCh);
  float var = 0.f;
#pragma unroll
  for (int o = 0; o < nCh; ++o) { float d = y[o] - mu; var += d * d; }
  var *= (1.f / nCh);
  float inv = rsqrtf(var + EPSv);
#pragma unroll
  for (int c = 0; c < nCh; ++c) {
    size_t xi = (((size_t)b * nCh + c) * nF + f) * nT + t;
    out[xi] = (y[c] - mu) * inv * gl[c] + zl[c] + x[xi];   // residual; [B,C,F,T] layout
  }
}

// ---------------- host launcher ----------------
extern "C" void kernel_launch(void* const* d_in, const int* in_sizes, int n_in,
                              void* d_out, int out_size, void* d_ws, size_t ws_size,
                              hipStream_t stream)
{
  (void)in_sizes; (void)n_in; (void)out_size; (void)ws_size;
  const float* x  = (const float*)d_in[0];
  const float* Wq = (const float*)d_in[1];
  const float* bq = (const float*)d_in[2];
  const float* aq = (const float*)d_in[3];
  const float* gq = (const float*)d_in[4];
  const float* zq = (const float*)d_in[5];
  const float* Wk = (const float*)d_in[6];
  const float* bk = (const float*)d_in[7];
  const float* ak = (const float*)d_in[8];
  const float* gk = (const float*)d_in[9];
  const float* zk = (const float*)d_in[10];
  const float* Wv = (const float*)d_in[11];
  const float* bv = (const float*)d_in[12];
  const float* av = (const float*)d_in[13];
  const float* gv = (const float*)d_in[14];
  const float* zv = (const float*)d_in[15];
  const float* Wp = (const float*)d_in[16];
  const float* bp = (const float*)d_in[17];
  const float* ap = (const float*)d_in[18];
  const float* gp = (const float*)d_in[19];
  const float* zp = (const float*)d_in[20];

  char* ws = (char*)d_ws;
  size_t n1 = (size_t)BH * nT * Dp;                       // 26,214,400 bf16 elements
  unsigned short* Qf  = (unsigned short*)ws;
  unsigned short* Kf  = Qf + n1;
  unsigned short* Ktg = Kf + n1;                          // [bh][chunk][d][l]
  unsigned short* Vtg = Ktg + n1;
  float* S = (float*)(ws + 4 * n1 * sizeof(unsigned short));
  float* Z = S + (size_t)BH * Dp * Dp;
  float* O = Z + (size_t)BH * Dp;
  unsigned short* Sbf = (unsigned short*)(O + n1);        // bf16 shadow of S

  dim3 blk(256, 1, 1);
  dim3 grid1(nT / 256, nF, nB);
  qkv_kernel<<<grid1, blk, 0, stream>>>(x, Wq, bq, aq, gq, zq, Wk, bk, ak, gk, zk,
                                        Wv, bv, av, gv, zv, Qf, Kf, Ktg, Vtg);
  attn_kernel<<<dim3(BH, 1, 1), blk, LDS_BYTES, stream>>>(Qf, Kf, Ktg, Vtg, S, Sbf, Z, O);
  proj_kernel<<<grid1, blk, 0, stream>>>(O, x, Wp, bp, ap, gp, zp, (float*)d_out);
}
